// DCNv4BlockLite_87926570484470
// MI455X (gfx1250) — matrix operand order repaired and checked by reading, not compiled, not run on hardware
//
#include <hip/hip_runtime.h>
#include <hip/hip_bf16.h>
#include <math.h>

// ---------------- problem constants (fixed by the reference) ----------------
#define Bc    4
#define Hc    64
#define Wc    64
#define Nc    (Hc * Wc)          // 4096
#define Cc    256
#define Gc    8
#define Cgc   (Cc / Gc)          // 32
#define Kpt   9                  // 3x3 taps
#define HIDc  (4 * Cc)           // 1024
#define OMraw (Gc * Kpt * 3)     // 216
#define OMS   224                // 216 padded to multiple of 32
#define Mrows (Bc * Nc)          // 16384 rows for every GEMM

typedef __attribute__((ext_vector_type(16))) _Float16 v16h;
typedef __attribute__((ext_vector_type(8)))  float    v8f;

// ---------------- weight conversion with transpose + pad ----------------
// in:  (K x Nsrc) row-major f32   out: (Npad x K) row-major f16 (N-major),
// rows n >= Nsrc are zero-filled.
__global__ void cvt_f16_T_kernel(const float* __restrict__ in,
                                 _Float16* __restrict__ out,
                                 int K, int Npad, int Nsrc) {
    int i = blockIdx.x * blockDim.x + threadIdx.x;
    if (i >= K * Npad) return;
    int n = i / K;
    int k = i - n * K;
    out[i] = (n < Nsrc) ? (_Float16)in[k * Nsrc + n] : (_Float16)0.0f;
}

// ---------------- LayerNorm (f32 in -> f16 out), one block per row ----------------
__global__ void ln_f16_kernel(const float* __restrict__ x,
                              const float* __restrict__ w,
                              const float* __restrict__ b,
                              _Float16* __restrict__ out) {
    __shared__ float red[Cc];
    const int row = blockIdx.x;
    const int tid = threadIdx.x;               // 256 threads == C
    const float xv = x[(size_t)row * Cc + tid];

    red[tid] = xv;
    __syncthreads();
    for (int s = Cc / 2; s > 0; s >>= 1) {
        if (tid < s) red[tid] += red[tid + s];
        __syncthreads();
    }
    const float mu = red[0] * (1.0f / Cc);
    __syncthreads();

    const float d = xv - mu;
    red[tid] = d * d;
    __syncthreads();
    for (int s = Cc / 2; s > 0; s >>= 1) {
        if (tid < s) red[tid] += red[tid + s];
        __syncthreads();
    }
    const float var = red[0] * (1.0f / Cc);
    const float rs  = rsqrtf(var + 1e-6f);
    out[(size_t)row * Cc + tid] = (_Float16)(d * rs * w[tid] + b[tid]);
}

// ---------------- WMMA GEMM: out = epi(A(f16, MxK) @ W(f16, KxN) + bias) ----------------
// Weights pre-transposed: WtT is (N x K) row-major -> both A and B fragments are
// contiguous 16-byte runs per lane (global_load_b128 clauses).
// block = 256 threads (8 waves); wave w computes a 32(M) x 32(N) tile:
//   m0 = by*32, n0 = bx*256 + w*32   (requires M % 32 == 0, N % 32 == 0, Kd % 64 == 0)
// 2x2 register tiling (4 WMMAs / 32-k step) + ping-pong software pipeline:
// two statically-indexed fragment sets alternate between "in use by WMMA" and
// "being prefetched", so there are no register copies and no spills.
// epilogue: 0 = f32 out ; 2 = GELU(exact) -> f16 out ; 3 = residual + gamma*val -> f32 out
__global__ void wmma_gemm_kernel(const _Float16* __restrict__ A,
                                 const _Float16* __restrict__ WtT,
                                 const float* __restrict__ bias, int biasN,
                                 const float* __restrict__ residual,
                                 const float* __restrict__ gamma,
                                 float* __restrict__ outF,
                                 _Float16* __restrict__ outH,
                                 int Kd, int N, int epilogue) {
    const int lane = threadIdx.x & 31;
    const int wave = threadIdx.x >> 5;
    const int n0 = blockIdx.x * 256 + wave * 32;
    if (n0 >= N) return;                       // wave-uniform
    const int m0 = blockIdx.y * 32;

    const int l15  = lane & 15;
    const int aKhi = (lane & 16) >> 1;         // +8 for upper half-wave (A layout)
    const int bKhi = (lane & 16) >> 1;         // +8 for upper half-wave (B layout)
    const _Float16* Arow0 = A   + (unsigned)(m0 + l15) * (unsigned)Kd;
    const _Float16* Arow1 = Arow0 + (unsigned)(16 * Kd);
    const _Float16* Bcol0 = WtT + (unsigned)(n0 + l15) * (unsigned)Kd;
    const _Float16* Bcol1 = Bcol0 + (unsigned)(16 * Kd);

    union frag { v16h v; _Float16 e[16]; };
    auto loadA = [&](frag& f, const _Float16* row, int k0) {
#pragma unroll
        for (int h = 0; h < 8; ++h) {
            f.e[h]     = row[k0 + aKhi + h];
            f.e[h + 8] = row[k0 + aKhi + 16 + h];
        }
    };
    auto loadB = [&](frag& f, const _Float16* col, int k0) {
        const int kb = k0 + bKhi;
#pragma unroll
        for (int h = 0; h < 8; ++h) { f.e[h] = col[kb + h]; f.e[h + 8] = col[kb + 16 + h]; }
    };

    v8f acc00 = {}, acc01 = {}, acc10 = {}, acc11 = {};
    frag a0s0, a1s0, b0s0, b1s0;   // set 0
    frag a0s1, a1s1, b0s1, b1s1;   // set 1

    loadA(a0s0, Arow0, 0);
    loadA(a1s0, Arow1, 0);
    loadB(b0s0, Bcol0, 0);
    loadB(b1s0, Bcol1, 0);

    for (int k0 = 0; k0 < Kd; k0 += 64) {
        // prefetch k0+32 into set 1 (always valid: Kd % 64 == 0)
        loadA(a0s1, Arow0, k0 + 32);
        loadA(a1s1, Arow1, k0 + 32);
        loadB(b0s1, Bcol0, k0 + 32);
        loadB(b1s1, Bcol1, k0 + 32);

        acc00 = __builtin_amdgcn_wmma_f32_16x16x32_f16(
            false, a0s0.v, false, b0s0.v, (short)0, acc00, false, false);
        acc01 = __builtin_amdgcn_wmma_f32_16x16x32_f16(
            false, a0s0.v, false, b1s0.v, (short)0, acc01, false, false);
        acc10 = __builtin_amdgcn_wmma_f32_16x16x32_f16(
            false, a1s0.v, false, b0s0.v, (short)0, acc10, false, false);
        acc11 = __builtin_amdgcn_wmma_f32_16x16x32_f16(
            false, a1s0.v, false, b1s0.v, (short)0, acc11, false, false);

        if (k0 + 64 < Kd) {                    // prefetch k0+64 into set 0
            loadA(a0s0, Arow0, k0 + 64);
            loadA(a1s0, Arow1, k0 + 64);
            loadB(b0s0, Bcol0, k0 + 64);
            loadB(b1s0, Bcol1, k0 + 64);
        }

        acc00 = __builtin_amdgcn_wmma_f32_16x16x32_f16(
            false, a0s1.v, false, b0s1.v, (short)0, acc00, false, false);
        acc01 = __builtin_amdgcn_wmma_f32_16x16x32_f16(
            false, a0s1.v, false, b1s1.v, (short)0, acc01, false, false);
        acc10 = __builtin_amdgcn_wmma_f32_16x16x32_f16(
            false, a1s1.v, false, b0s1.v, (short)0, acc10, false, false);
        acc11 = __builtin_amdgcn_wmma_f32_16x16x32_f16(
            false, a1s1.v, false, b1s1.v, (short)0, acc11, false, false);
    }

    // C/D layout: VGPR i -> row tile_m0 + i + ((lane&16)?8:0), col = tile_n0 + l15
    const int rb = (lane & 16) >> 1;
#pragma unroll
    for (int mt = 0; mt < 2; ++mt) {
#pragma unroll
        for (int nt = 0; nt < 2; ++nt) {
            const v8f acc = (mt == 0) ? ((nt == 0) ? acc00 : acc01)
                                      : ((nt == 0) ? acc10 : acc11);
            const int ccol = n0 + nt * 16 + l15;
            const float bsv = (ccol < biasN) ? bias[ccol] : 0.0f;
            const int rbase = m0 + mt * 16 + rb;
#pragma unroll
            for (int i = 0; i < 8; ++i) {
                const unsigned idx = (unsigned)(rbase + i) * (unsigned)N + (unsigned)ccol;
                const float val = acc[i] + bsv;
                if (epilogue == 0) {
                    outF[idx] = val;
                } else if (epilogue == 2) {
                    const float gel = 0.5f * val * (1.0f + erff(val * 0.70710678118654752f));
                    outH[idx] = (_Float16)gel;
                } else { // 3: residual + gamma * val
                    outF[idx] = residual[idx] + gamma[ccol] * val;
                }
            }
        }
    }
}

// ---------------- DCNv4 sampling: one wave per (b, p, g); lane = channel in group ----------------
__global__ void dcn_sample_kernel(const float* __restrict__ v,
                                  const float* __restrict__ om,
                                  _Float16* __restrict__ dcn) {
    const int wid = blockIdx.x * (blockDim.x >> 5) + (threadIdx.x >> 5);
    if (wid >= Bc * Nc * Gc) return;
    const int lane = threadIdx.x & 31;
    const int g = wid & (Gc - 1);
    const int p = (wid >> 3) & (Nc - 1);
    const int b = wid >> 15;
    const int py = p >> 6;                     // W=64
    const int px = p & 63;

    const float* vb  = v  + (unsigned)b * (Nc * Cc) + g * Cgc + lane;
    const float* omp = om + (unsigned)(b * Nc + p) * OMS + g * (Kpt * 3);

    float acc = 0.0f;
#pragma unroll
    for (int k = 0; k < Kpt; ++k) {
        const float dy = omp[k * 3 + 0];
        const float dx = omp[k * 3 + 1];
        const float mk = omp[k * 3 + 2];
        const float ys = (float)(py + (k / 3) - 1) + dy;
        const float xs = (float)(px + (k % 3) - 1) + dx;
        const float y0f = floorf(ys), x0f = floorf(xs);
        const int y0 = (int)y0f, x0 = (int)x0f;
        const float wy1 = ys - y0f, wy0 = 1.0f - wy1;
        const float wx1 = xs - x0f, wx0 = 1.0f - wx1;
        float s = 0.0f;
        {
            const int yy = y0, xx = x0;
            if (yy >= 0 && yy < Hc && xx >= 0 && xx < Wc)
                s += wy0 * wx0 * vb[(unsigned)(yy * Wc + xx) * Cc];
        }
        {
            const int yy = y0, xx = x0 + 1;
            if (yy >= 0 && yy < Hc && xx >= 0 && xx < Wc)
                s += wy0 * wx1 * vb[(unsigned)(yy * Wc + xx) * Cc];
        }
        {
            const int yy = y0 + 1, xx = x0;
            if (yy >= 0 && yy < Hc && xx >= 0 && xx < Wc)
                s += wy1 * wx0 * vb[(unsigned)(yy * Wc + xx) * Cc];
        }
        {
            const int yy = y0 + 1, xx = x0 + 1;
            if (yy >= 0 && yy < Hc && xx >= 0 && xx < Wc)
                s += wy1 * wx1 * vb[(unsigned)(yy * Wc + xx) * Cc];
        }
        acc += mk * s;
    }
    dcn[(unsigned)(b * Nc + p) * Cc + g * Cgc + lane] = (_Float16)acc;
}

// ---------------- host ----------------
extern "C" void kernel_launch(void* const* d_in, const int* in_sizes, int n_in,
                              void* d_out, int out_size, void* d_ws, size_t ws_size,
                              hipStream_t stream) {
    const float* x       = (const float*)d_in[0];
    const float* norm1_w = (const float*)d_in[1];
    const float* norm1_b = (const float*)d_in[2];
    const float* Wv      = (const float*)d_in[3];
    const float* bv      = (const float*)d_in[4];
    const float* Woff    = (const float*)d_in[5];
    const float* boff    = (const float*)d_in[6];
    const float* Wo      = (const float*)d_in[7];
    const float* bo      = (const float*)d_in[8];
    const float* gamma1  = (const float*)d_in[9];
    const float* norm2_w = (const float*)d_in[10];
    const float* norm2_b = (const float*)d_in[11];
    const float* W1      = (const float*)d_in[12];
    const float* b1      = (const float*)d_in[13];
    const float* W2      = (const float*)d_in[14];
    const float* b2      = (const float*)d_in[15];
    const float* gamma2  = (const float*)d_in[16];
    float* out = (float*)d_out;

    char* ws = (char*)d_ws;
    size_t off = 0;
    auto carve = [&](size_t bytes) -> void* {
        void* p = ws + off;
        off = (off + bytes + 255) & ~(size_t)255;
        return p;
    };
    _Float16* yh    = (_Float16*)carve((size_t)Mrows * Cc * 2);
    _Float16* WvT   = (_Float16*)carve((size_t)Cc * Cc * 2);      // (256 x 256) N-major
    _Float16* WoffT = (_Float16*)carve((size_t)OMS * Cc * 2);     // (224 x 256) N-major
    _Float16* WoT   = (_Float16*)carve((size_t)Cc * Cc * 2);
    _Float16* W1T   = (_Float16*)carve((size_t)HIDc * Cc * 2);    // (1024 x 256)
    _Float16* W2T   = (_Float16*)carve((size_t)Cc * HIDc * 2);    // (256 x 1024)
    float*    vbuf  = (float*)   carve((size_t)Mrows * Cc * 4);
    float*    ombuf = (float*)   carve((size_t)Mrows * OMS * 4);
    _Float16* dcnH  = (_Float16*)carve((size_t)Mrows * Cc * 2);
    float*    x1    = (float*)   carve((size_t)Mrows * Cc * 4);
    _Float16* y2h   = (_Float16*)carve((size_t)Mrows * Cc * 2);
    _Float16* hh    = (_Float16*)carve((size_t)Mrows * HIDc * 2);
    (void)ws_size;

    // 1) weight conversions: f32 row-major (K x N) -> f16 transposed (Npad x K)
    cvt_f16_T_kernel<<<(Cc * Cc + 255) / 256, 256, 0, stream>>>(Wv, WvT, Cc, Cc, Cc);
    cvt_f16_T_kernel<<<(OMS * Cc + 255) / 256, 256, 0, stream>>>(Woff, WoffT, Cc, OMS, OMraw);
    cvt_f16_T_kernel<<<(Cc * Cc + 255) / 256, 256, 0, stream>>>(Wo, WoT, Cc, Cc, Cc);
    cvt_f16_T_kernel<<<(HIDc * Cc + 255) / 256, 256, 0, stream>>>(W1, W1T, Cc, HIDc, HIDc);
    cvt_f16_T_kernel<<<(Cc * HIDc + 255) / 256, 256, 0, stream>>>(W2, W2T, HIDc, Cc, Cc);

    // 2) LN1: x -> yh (f16)
    ln_f16_kernel<<<Mrows, Cc, 0, stream>>>(x, norm1_w, norm1_b, yh);

    // 3) value projection: v = yh @ Wv + bv   (f32 out)
    wmma_gemm_kernel<<<dim3((Cc + 255) / 256, Mrows / 32), 256, 0, stream>>>(
        yh, WvT, bv, Cc, nullptr, nullptr, vbuf, nullptr, Cc, Cc, 0);

    // 4) offset/mask projection: om = yh @ Woff + boff   (f32 out, N padded to 224)
    wmma_gemm_kernel<<<dim3((OMS + 255) / 256, Mrows / 32), 256, 0, stream>>>(
        yh, WoffT, boff, OMraw, nullptr, nullptr, ombuf, nullptr, Cc, OMS, 0);

    // 5) deformable sampling + modulation -> dcnH (f16)
    dcn_sample_kernel<<<(Bc * Nc * Gc) / 8, 256, 0, stream>>>(vbuf, ombuf, dcnH);

    // 6) output projection + residual*gamma1: x1 = x + gamma1*(dcnH @ Wo + bo)
    wmma_gemm_kernel<<<dim3((Cc + 255) / 256, Mrows / 32), 256, 0, stream>>>(
        dcnH, WoT, bo, Cc, x, gamma1, x1, nullptr, Cc, Cc, 3);

    // 7) LN2: x1 -> y2h (f16)
    ln_f16_kernel<<<Mrows, Cc, 0, stream>>>(x1, norm2_w, norm2_b, y2h);

    // 8) MLP up + exact GELU: hh = gelu(y2h @ W1 + b1)  (f16 out)
    wmma_gemm_kernel<<<dim3((HIDc + 255) / 256, Mrows / 32), 256, 0, stream>>>(
        y2h, W1T, b1, HIDc, nullptr, nullptr, nullptr, hh, Cc, HIDc, 2);

    // 9) MLP down + residual*gamma2 -> d_out: out = x1 + gamma2*(hh @ W2 + b2)
    wmma_gemm_kernel<<<dim3((Cc + 255) / 256, Mrows / 32), 256, 0, stream>>>(
        hh, W2T, b2, Cc, x1, gamma2, out, nullptr, HIDc, Cc, 3);
}
